// RecurrentDecoder_3736621547807
// MI455X (gfx1250) — matrix operand order repaired and checked by reading, not compiled, not run on hardware
//
#include <hip/hip_runtime.h>
#include <hip/hip_bf16.h>
#include <math.h>

// ---------------------------------------------------------------------------
// GRU decoder, MI455X (gfx1250) persistent-cluster design.
//   B=256, H=512, F=64, T=1024.
// Fused recurrence:  g = h @ W_bigT + b_big,  W_big has 4 groups of 512 rows:
//   g0: W_hh_r + W_ih_r@W_dense   (fused r pre-activation)
//   g1: W_hh_z + W_ih_z@W_dense   (fused z pre-activation)
//   g2: W_ih_n@W_dense            (i_n, input side of n gate)
//   g3: W_hh_n                    (h_n, hidden side of n gate)
//   n = tanh(i_n + b_in + r*(h_n + b_hn));  h' = (1-z)*n + z*h
// Output t = h_{t+1} @ W_dense^T + b_dense, emitted one iteration late from
// the globally broadcast h fragments.
//
// Round-2 change: per-wave GEMM runs as two sequential 1-Mtile passes
// (acc = 8 v8f = 64 VGPRs live instead of 128) to eliminate the scratch
// spills seen in round-1 disasm; gate biases hoisted out of the t-loop.
// ---------------------------------------------------------------------------

typedef __attribute__((ext_vector_type(16))) __bf16 v16bf;
typedef __attribute__((ext_vector_type(8)))  float  v8f;

#define B_SZ   256
#define H_SZ   512
#define F_SZ   64
#define T_SZ   1024
#define NWG    16
#define TPB    256

// fragment element counts
#define HFRAG_N   (16 * 16 * 32 * 16)      // [mt(16)][kb(16)][lane(32)][e(16)]
#define LDSW_N    (8  * 16 * 32 * 16)      // [ct(8)][kb(16)][lane(32)][e(16)]
#define LDSWD_N   (4  * 16 * 32 * 16)      // [ft(4)][kb(16)][lane(32)][e(16)]

// workspace byte layout
#define WBIG_BYTES   (2048u * 512u * 4u)               // 4 MB f32 fused weights
#define BBIG_BYTES   (2048u * 4u)
#define HBUF_BYTES   (2u * HFRAG_N * 2u)               // double-buffered bf16 frags
#define WS_WBIG      0u
#define WS_BBIG      (WS_WBIG + WBIG_BYTES)
#define WS_HBUF      (WS_BBIG + BBIG_BYTES)
#define WS_BAR       (WS_HBUF + HBUF_BYTES)
#define WS_TOTAL     (WS_BAR + 256u)

// ---- bf16 + WMMA fragment helpers (ISA 7.12.2 layouts) --------------------

__device__ __forceinline__ unsigned short f2bf(float x) {
  unsigned u = __builtin_bit_cast(unsigned, x);
  unsigned r = (u + 0x7FFFu + ((u >> 16) & 1u)) >> 16;   // round-to-nearest-even
  return (unsigned short)r;
}

// A-matrix 16x32 bf16: lanes 0-15 -> M=lane, K = {0..7,16..23}; lanes 16-31 ->
// M=lane-16, K = {8..15,24..31} (2 halfs per VGPR, 8 VGPRs).
__device__ __forceinline__ int frag_k_a(int lane, int e) {
  return ((lane & 16) ? 8 : 0) + ((e < 8) ? e : (e + 8));
}
// B-matrix 32x16 bf16: lanes 0-15 -> N=lane, K = {0..7,16..23}; lanes 16-31 ->
// N=lane-16, K = {8..15,24..31} (same K order as A).
__device__ __forceinline__ int frag_k_b(int lane, int e) {
  return ((lane & 16) ? 8 : 0) + ((e < 8) ? e : (e + 8));
}

__device__ __forceinline__ v8f wmma_bf16(v16bf a, v16bf b, v8f c) {
  return __builtin_amdgcn_wmma_f32_16x16x32_bf16(false, a, false, b,
                                                 (short)0, c, false, false);
}

__device__ __forceinline__ float sigmoidf_(float x) {
  return 1.0f / (1.0f + __expf(-x));
}

// ---- prep kernels ---------------------------------------------------------

__global__ void prep_wbig(const float* __restrict__ Wih,
                          const float* __restrict__ Whh,
                          const float* __restrict__ Wd,
                          float* __restrict__ Wbig) {
  int idx = blockIdx.x * blockDim.x + threadIdx.x;     // 2048*512
  if (idx >= 2048 * 512) return;
  int row = idx >> 9;          // 0..2047
  int k   = idx & 511;
  int g   = row >> 9;          // group 0..3
  int j   = row & 511;
  float v;
  if (g <= 1) {                // fused r / z: W_hh + W_ih @ W_dense
    int n = g * 512 + j;
    float s = 0.f;
    #pragma unroll 8
    for (int f = 0; f < F_SZ; ++f) s += Wih[n * 64 + f] * Wd[f * 512 + k];
    v = Whh[n * 512 + k] + s;
  } else if (g == 2) {         // i_n side: W_ih_n @ W_dense
    int n = 1024 + j;
    float s = 0.f;
    #pragma unroll 8
    for (int f = 0; f < F_SZ; ++f) s += Wih[n * 64 + f] * Wd[f * 512 + k];
    v = s;
  } else {                     // h_n side: W_hh_n
    v = Whh[(1024 + j) * 512 + k];
  }
  Wbig[idx] = v;
}

__global__ void prep_bbig(const float* __restrict__ Wih,
                          const float* __restrict__ bih,
                          const float* __restrict__ bhh,
                          const float* __restrict__ bd,
                          float* __restrict__ bbig,
                          int* __restrict__ bar) {
  int idx = blockIdx.x * blockDim.x + threadIdx.x;
  if (idx < 4) bar[idx] = 0;                    // reset software barrier state
  if (idx >= 2048) return;
  int g = idx >> 9, j = idx & 511;
  float v;
  if (g <= 1) {
    int n = g * 512 + j;
    float s = 0.f;
    for (int f = 0; f < F_SZ; ++f) s += Wih[n * 64 + f] * bd[f];
    v = bih[n] + bhh[n] + s;
  } else if (g == 2) {
    int n = 1024 + j;
    float s = 0.f;
    for (int f = 0; f < F_SZ; ++f) s += Wih[n * 64 + f] * bd[f];
    v = bih[n] + s;
  } else {
    v = bhh[1024 + j];
  }
  bbig[idx] = v;
}

__global__ void pack_h0(const float* __restrict__ h0,
                        unsigned short* __restrict__ hbuf0) {
  int idx = blockIdx.x * blockDim.x + threadIdx.x;     // HFRAG_N
  if (idx >= HFRAG_N) return;
  int e = idx & 15, lane = (idx >> 4) & 31, kb = (idx >> 9) & 15, mt = idx >> 13;
  int m = 16 * mt + (lane & 15);
  int k = 32 * kb + frag_k_a(lane, e);
  hbuf0[idx] = f2bf(h0[m * H_SZ + k]);
}

// ---- persistent recurrent kernel ------------------------------------------

__device__ __forceinline__ void dense_out_tile(
    const unsigned short* __restrict__ hc,      // h fragments [mt][kb][lane][e]
    const unsigned short* __restrict__ lds_wd,  // W_dense B fragments
    float bias,
    float* __restrict__ out,
    int w, int wv, int lane, int tcol) {
  v8f c = {0.f, 0.f, 0.f, 0.f, 0.f, 0.f, 0.f, 0.f};
  #pragma unroll
  for (int kb = 0; kb < 16; ++kb) {
    v16bf a = *(const v16bf*)(hc + (((w * 16) + kb) * 32 + lane) * 16);
    v16bf b = *(const v16bf*)(lds_wd + (((wv * 16) + kb) * 32 + lane) * 16);
    c = wmma_bf16(a, b, c);
  }
  int f = 16 * wv + (lane & 15);
  #pragma unroll
  for (int r = 0; r < 8; ++r) {
    int m = 16 * w + r + 8 * (lane >> 4);   // C-tile layout (ISA 7.12.2)
    out[((size_t)m * T_SZ + tcol) * F_SZ + f] = c[r] + bias;
  }
}

__global__ void gru_persistent(const float* __restrict__ Wbig,
                               const float* __restrict__ bbig,
                               const float* __restrict__ Wd,
                               const float* __restrict__ bd,
                               const float* __restrict__ h0,
                               unsigned short* __restrict__ hbuf,
                               float* __restrict__ out,
                               int* __restrict__ bar,
                               int T, int use_cluster) {
  extern __shared__ char smem[];
  unsigned short* lds_w  = (unsigned short*)smem;            // 128 KB weights
  unsigned short* lds_wd = lds_w + LDSW_N;                   //  64 KB W_dense
  float*          lds_h  = (float*)(lds_wd + LDSWD_N);       //  32 KB own h slice

  const int w    = blockIdx.x;          // WGP id = hidden slice [32w, 32w+32)
  const int tid  = threadIdx.x;
  const int lane = tid & 31;
  const int wv   = tid >> 5;            // 8 waves

  // --- prologue: pack weights into LDS (resident for all 1024 steps) -------
  for (int i = tid; i < LDSW_N; i += TPB) {
    int e = i & 15, l = (i >> 4) & 31, kb = (i >> 9) & 15, ct = i >> 13;
    int g   = ct >> 1;
    int row = g * 512 + 32 * w + 16 * (ct & 1) + (l & 15);
    int k   = 32 * kb + frag_k_b(l, e);
    lds_w[i] = f2bf(Wbig[row * 512 + k]);
  }
  for (int i = tid; i < LDSWD_N; i += TPB) {
    int e = i & 15, l = (i >> 4) & 31, kb = (i >> 9) & 15, ft = i >> 13;
    int f = 16 * ft + (l & 15);
    int k = 32 * kb + frag_k_b(l, e);
    lds_wd[i] = f2bf(Wd[f * 512 + k]);
  }
  for (int i = tid; i < B_SZ * 32; i += TPB) {
    int m = i >> 5, jl = i & 31;
    lds_h[i] = h0[m * H_SZ + 32 * w + jl];
  }
  __syncthreads();

  // gate biases are invariant over t: hoist (per-thread, indexed by lane col)
  const int nl = lane & 15;
  float brc[2], bzc[2], bic[2], bhc[2];
  #pragma unroll
  for (int p = 0; p < 2; ++p) {
    int jl = 16 * p + nl;
    brc[p] = bbig[0 * 512 + 32 * w + jl];
    bzc[p] = bbig[1 * 512 + 32 * w + jl];
    bic[p] = bbig[2 * 512 + 32 * w + jl];
    bhc[p] = bbig[3 * 512 + 32 * w + jl];
  }
  const float bdense = (wv < 4) ? bd[16 * wv + nl] : 0.f;

  // --- recurrence ----------------------------------------------------------
  for (int t = 0; t < T; ++t) {
    const unsigned short* hc = hbuf + (size_t)(t & 1) * HFRAG_N;   // h_t frags

    // emit out[:, t-1, :] = dense(h_t) (one iteration late; waves 0-3)
    if (t > 0 && wv < 4)
      dense_out_tile(hc, lds_wd, bdense, out, w, wv, lane, t - 1);

    // gate GEMM + gating: two sequential passes, 1 M-tile x 8 column tiles
    // each (r,r,z,z,in,in,hn,hn) -> only 8 f32 accumulator tiles live.
    #pragma unroll 1
    for (int mi = 0; mi < 2; ++mi) {
      const int mt = 2 * wv + mi;
      v8f acc[8];
      {
        v8f z = {0.f, 0.f, 0.f, 0.f, 0.f, 0.f, 0.f, 0.f};
        #pragma unroll
        for (int ct = 0; ct < 8; ++ct) acc[ct] = z;
      }
      for (int kb = 0; kb < 16; ++kb) {
        v16bf a = *(const v16bf*)(hc + (((mt * 16) + kb) * 32 + lane) * 16);
        #pragma unroll
        for (int ct = 0; ct < 8; ++ct) {
          v16bf b = *(const v16bf*)(lds_w + (((ct * 16) + kb) * 32 + lane) * 16);
          acc[ct] = wmma_bf16(a, b, acc[ct]);
        }
      }
      // gating (VALU), in-place update of the WGP's f32 h slice in LDS
      #pragma unroll
      for (int p = 0; p < 2; ++p) {
        int jl = 16 * p + nl;
        #pragma unroll
        for (int r = 0; r < 8; ++r) {
          int m = 16 * mt + r + 8 * (lane >> 4);
          float rg = sigmoidf_(acc[0 + p][r] + brc[p]);
          float zg = sigmoidf_(acc[2 + p][r] + bzc[p]);
          float ng = tanhf((acc[4 + p][r] + bic[p]) + rg * (acc[6 + p][r] + bhc[p]));
          float ho = lds_h[m * 32 + jl];
          lds_h[m * 32 + jl] = (1.f - zg) * ng + zg * ho;
        }
      }
    }
    __syncthreads();

    // broadcast: repack our 32-wide K-block (kb = w) of h_{t+1} as A fragments
    unsigned short* hn = hbuf + (size_t)((t + 1) & 1) * HFRAG_N;
    for (int i = tid; i < 16 * 32 * 16; i += TPB) {
      int e = i & 15, l = (i >> 4) & 31, mt2 = i >> 9;
      int m = 16 * mt2 + (l & 15);
      int k = frag_k_a(l, e);                       // local K within our block
      hn[(((mt2 * 16) + w) * 32 + l) * 16 + e] = f2bf(lds_h[m * 32 + k]);
    }
    __threadfence();                                 // device-scope visibility

    // cross-WGP sync: CDNA5 cluster barrier when dispatched as a cluster,
    // otherwise a device-scope atomic barrier over the 16 persistent WGs.
    if (use_cluster) {
      __builtin_amdgcn_s_cluster_barrier();          // s_barrier_signal/wait -3
    } else {
      __syncthreads();
      if (tid == 0) {
        int gen = __hip_atomic_load(&bar[1], __ATOMIC_RELAXED,
                                    __HIP_MEMORY_SCOPE_AGENT);
        if (__hip_atomic_fetch_add(&bar[0], 1, __ATOMIC_ACQ_REL,
                                   __HIP_MEMORY_SCOPE_AGENT) == NWG - 1) {
          __hip_atomic_store(&bar[0], 0, __ATOMIC_RELAXED,
                             __HIP_MEMORY_SCOPE_AGENT);
          __hip_atomic_fetch_add(&bar[1], 1, __ATOMIC_RELEASE,
                                 __HIP_MEMORY_SCOPE_AGENT);
        } else {
          while (__hip_atomic_load(&bar[1], __ATOMIC_ACQUIRE,
                                   __HIP_MEMORY_SCOPE_AGENT) == gen)
            __builtin_amdgcn_s_sleep(1);
        }
      }
      __syncthreads();
    }
  }

  // epilogue: out[:, T-1, :] = dense(h_T)
  if (wv < 4) {
    const unsigned short* hc = hbuf + (size_t)(T & 1) * HFRAG_N;
    dense_out_tile(hc, lds_wd, bdense, out, w, wv, lane, T - 1);
  }
}

// ---- host launcher --------------------------------------------------------

extern "C" void kernel_launch(void* const* d_in, const int* in_sizes, int n_in,
                              void* d_out, int out_size, void* d_ws, size_t ws_size,
                              hipStream_t stream) {
  (void)in_sizes; (void)n_in; (void)out_size;
  const float* h0  = (const float*)d_in[0];
  const float* Wih = (const float*)d_in[1];
  const float* Whh = (const float*)d_in[2];
  const float* bih = (const float*)d_in[3];
  const float* bhh = (const float*)d_in[4];
  const float* Wd  = (const float*)d_in[5];
  const float* bd  = (const float*)d_in[6];
  float* out = (float*)d_out;

  if (ws_size < (size_t)WS_TOTAL) return;   // deterministic no-op guard
  char* ws = (char*)d_ws;
  float*          Wbig = (float*)(ws + WS_WBIG);
  float*          bbig = (float*)(ws + WS_BBIG);
  unsigned short* hbuf = (unsigned short*)(ws + WS_HBUF);
  int*            bar  = (int*)(ws + WS_BAR);

  prep_wbig<<<(2048 * 512 + TPB - 1) / TPB, TPB, 0, stream>>>(Wih, Whh, Wd, Wbig);
  prep_bbig<<<(2048 + TPB - 1) / TPB, TPB, 0, stream>>>(Wih, bih, bhh, bd, bbig, bar);
  pack_h0<<<HFRAG_N / TPB, TPB, 0, stream>>>(h0, hbuf);

  size_t shmem = (size_t)(LDSW_N + LDSWD_N) * sizeof(unsigned short)
               + (size_t)B_SZ * 32 * sizeof(float);          // 224 KB < 320 KB LDS
  (void)hipFuncSetAttribute((const void*)gru_persistent,
                            hipFuncAttributeMaxDynamicSharedMemorySize,
                            (int)shmem);
  gru_persistent<<<NWG, TPB, shmem, stream>>>(Wbig, bbig, Wd, bd, h0, hbuf, out,
                                              bar, T_SZ, /*use_cluster=*/0);
}
